// BilinearAttention_8229157339152
// MI455X (gfx1250) — matrix operand order repaired and checked by reading, not compiled, not run on hardware
//
#include <hip/hip_runtime.h>
#include <hip/hip_bf16.h>

// ---------------------------------------------------------------------------
// BilinearAttention (non-local block) for MI455X / gfx1250, wave32 + WMMA.
// B=4, C=256, H=W=64 -> HW=4096, d_qk=32, d_v=128.
// Flash-attention style fusion: the 4096x4096 attention matrix is never
// materialized (would be 268MB). All GEMMs use v_wmma_f32_16x16x32_f16.
// ---------------------------------------------------------------------------

#define NB   4
#define CC   256
#define HW   4096
#define DQK  32
#define DV   128
#define BN_EPS 1e-5f

typedef __attribute__((ext_vector_type(16))) _Float16 v16h;
typedef __attribute__((ext_vector_type(8)))  _Float16 v8h;
typedef __attribute__((ext_vector_type(8)))  float    v8f;

__device__ __forceinline__ v16h ld16(const _Float16* p) { return *(const v16h*)p; }

// Assemble a 16-half fragment from two contiguous 8-half chunks (A-operand).
__device__ __forceinline__ v16h ld8x2(const _Float16* p0, const _Float16* p1) {
    v8h a = *(const v8h*)p0;
    v8h b = *(const v8h*)p1;
    v16h r;
#pragma unroll
    for (int i = 0; i < 8; ++i) { r[i] = a[i]; r[i + 8] = b[i]; }
    return r;
}

__device__ __forceinline__ v8f wmma16x16x32(v16h a, v16h b, v8f c) {
    return __builtin_amdgcn_wmma_f32_16x16x32_f16(false, a, false, b,
                                                  (short)0, c, false, false);
}

// Reductions within the 16-lane half-groups of a wave32 (xor masks 1..8 never
// cross the lane<16 / lane>=16 boundary, matching the WMMA C/D row layout).
__device__ __forceinline__ float redmax16(float v) {
#pragma unroll
    for (int m = 1; m < 16; m <<= 1) v = fmaxf(v, __shfl_xor(v, m, 32));
    return v;
}
__device__ __forceinline__ float redsum16(float v) {
#pragma unroll
    for (int m = 1; m < 16; m <<= 1) v += __shfl_xor(v, m, 32);
    return v;
}

// ---------------------------------------------------------------------------
// Kernel 0a: fold BN into conv weights: wf[o][c] = w[o][c]*gamma/sqrt(var+eps)
//            t[o] = (b-mean)*s + beta
// ---------------------------------------------------------------------------
__global__ void fold_bn(const float* __restrict__ w, const float* __restrict__ bb,
                        const float* __restrict__ gamma, const float* __restrict__ beta,
                        const float* __restrict__ mean, const float* __restrict__ var,
                        _Float16* __restrict__ wf, float* __restrict__ t,
                        int Dout, int Cin) {
    int i = blockIdx.x * blockDim.x + threadIdx.x;
    int total = Dout * Cin;
    if (i < total) {
        int o = i / Cin;
        float s = gamma[o] * rsqrtf(var[o] + BN_EPS);
        wf[i] = (_Float16)(w[i] * s);
    }
    if (i < Dout) {
        float s = gamma[i] * rsqrtf(var[i] + BN_EPS);
        t[i] = (bb[i] - mean[i]) * s + beta[i];
    }
}

// ---------------------------------------------------------------------------
// Kernel 0b: transpose+convert [B][C][HW] f32 -> [B][HW][C] f16 (B-operand
// layout for the projection GEMMs: fixed n per lane, contiguous c).
// ---------------------------------------------------------------------------
__global__ void xpose_h(const float* __restrict__ in, _Float16* __restrict__ out) {
    long i = (long)blockIdx.x * blockDim.x + threadIdx.x;
    const long total = (long)NB * CC * HW;
    if (i >= total) return;
    int c = (int)(i & (CC - 1));
    long r = i >> 8;              // CC = 256 = 2^8
    int n = (int)(r & (HW - 1));
    long b = r >> 12;             // HW = 4096 = 2^12
    out[i] = (_Float16)in[(b * CC + c) * HW + n];
}

// ---------------------------------------------------------------------------
// Kernel 1: projection GEMM + folded BN:  out[o][n] = sum_c wf[o][c]*xh[n][c] + t[o]
//   mode 0: store transposed rows  outNT[b][n][Dout]   (theta/phi: Q/K layout)
//   mode 1: store natural          outCN[b][Dout][HW]  (g: V layout)
// Block = 128 threads = 4 waves; wave w owns a 16-n sub-tile; o-tile from z.
// ---------------------------------------------------------------------------
__global__ void proj_gemm(const _Float16* __restrict__ xh,  // [B][HW][CC]
                          const _Float16* __restrict__ wf,  // [Dout][CC]
                          const float* __restrict__ t,      // [Dout]
                          _Float16* __restrict__ outNT,
                          _Float16* __restrict__ outCN,
                          int Dout, int mode) {
    const int lane = threadIdx.x & 31, wave = threadIdx.x >> 5;
    const int b = blockIdx.x, nblk = blockIdx.y, oT = blockIdx.z;
    const int n0 = nblk * 64 + wave * 16;
    const int o0 = oT * 16;
    const int half = lane >> 4, l16 = lane & 15;

    const _Float16* xrow = xh + ((long)b * HW + (n0 + l16)) * CC;
    const _Float16* wrow = wf + (long)(o0 + l16) * CC;

    v8f acc = {};
#pragma unroll
    for (int cb = 0; cb < CC; cb += 32) {
        v16h a  = ld8x2(wrow + cb + half * 16, wrow + cb + half * 16 + 8);
        v16h bf = ld16(xrow + cb + half * 16);
        acc = wmma16x16x32(a, bf, acc);
    }

    if (mode == 0) {
        v8h r;
#pragma unroll
        for (int v = 0; v < 8; ++v) {
            int o = o0 + half * 8 + v;
            r[v] = (_Float16)(acc[v] + t[o]);
        }
        *(v8h*)(outNT + ((long)b * HW + (n0 + l16)) * Dout + o0 + half * 8) = r;
    } else {
#pragma unroll
        for (int v = 0; v < 8; ++v) {
            int o = o0 + half * 8 + v;
            outCN[((long)b * Dout + o) * HW + n0 + l16] = (_Float16)(acc[v] + t[o]);
        }
    }
}

// ---------------------------------------------------------------------------
// Kernel 2: fused flash attention.
//   attn[n,m] = softmax_m( theta[:,n].phi[:,m] ),  out[n,c] = sum_m attn*V[m,c]
// Block = 128 threads = 4 waves; each wave owns 16 query rows + O[16x128] in
// 64 VGPRs. Online softmax; P tile staged through 1KB/wave LDS to convert the
// WMMA C-layout into the A-layout for the PV matmuls.
// ---------------------------------------------------------------------------
__global__ void attn_kernel(const _Float16* __restrict__ thetaT, // [B][HW][32]
                            const _Float16* __restrict__ phiT,   // [B][HW][32]
                            const _Float16* __restrict__ gh,     // [B][128][HW]
                            _Float16* __restrict__ outT) {       // [B][HW][128]
    __shared__ __align__(16) _Float16 Pl[4][16][32];

    const int lane = threadIdx.x & 31, wave = threadIdx.x >> 5;
    const int b = blockIdx.x, nblk = blockIdx.y;
    const int n0 = nblk * 64 + wave * 16;
    const int half = lane >> 4, l16 = lane & 15;

    // Q fragment (A: 16n x 32d), loaded once, reused for all 128 m-blocks.
    const _Float16* qrow = thetaT + ((long)b * HW + n0 + l16) * DQK;
    const v16h qa = ld8x2(qrow + half * 16, qrow + half * 16 + 8);

    v8f O[8];
#pragma unroll
    for (int i = 0; i < 8; ++i) O[i] = (v8f){};
    float rmax[8], rsum[8];
#pragma unroll
    for (int v = 0; v < 8; ++v) { rmax[v] = -1e30f; rsum[v] = 0.f; }

    const _Float16* phib = phiT + (long)b * HW * DQK;
    const _Float16* gb   = gh   + (long)b * DV * HW;

    for (int m0 = 0; m0 < HW; m0 += 32) {
        // ---- scores: S[16n x 32m] as two 16x16 WMMA tiles
        v16h kb0 = ld16(phib + (long)(m0 + l16) * DQK + half * 16);
        v16h kb1 = ld16(phib + (long)(m0 + 16 + l16) * DQK + half * 16);
        v8f z = {};
        v8f s0 = wmma16x16x32(qa, kb0, z);
        v8f s1 = wmma16x16x32(qa, kb1, z);

        // ---- online softmax (row r = half*8+v lives on 16 lanes of one half)
        float p0[8], p1[8];
#pragma unroll
        for (int v = 0; v < 8; ++v) {
            float tm = redmax16(fmaxf(s0[v], s1[v]));
            float nm = fmaxf(rmax[v], tm);
            float alpha = __expf(rmax[v] - nm);
            p0[v] = __expf(s0[v] - nm);
            p1[v] = __expf(s1[v] - nm);
            float ts = redsum16(p0[v] + p1[v]);
            rsum[v] = rsum[v] * alpha + ts;
            rmax[v] = nm;
#pragma unroll
            for (int ct = 0; ct < 8; ++ct) O[ct][v] *= alpha;
        }

        // ---- C-layout -> A-layout via per-wave LDS tile
        __syncthreads();   // WAR: previous iteration's A-reads done
#pragma unroll
        for (int v = 0; v < 8; ++v) {
            Pl[wave][half * 8 + v][l16]      = (_Float16)p0[v];
            Pl[wave][half * 8 + v][16 + l16] = (_Float16)p1[v];
        }
        __syncthreads();   // RAW: P tile visible
        v16h pa = ld8x2(&Pl[wave][l16][half * 16], &Pl[wave][l16][half * 16 + 8]);

        // ---- O += P @ V  (8 c-tiles of 16)
#pragma unroll
        for (int ct = 0; ct < 8; ++ct) {
            v16h vb = ld16(gb + (long)(ct * 16 + l16) * HW + m0 + half * 16);
            O[ct] = wmma16x16x32(pa, vb, O[ct]);
        }
    }

    // ---- normalize and store out^T [n][128] f16 (B-operand layout for final GEMM)
#pragma unroll
    for (int v = 0; v < 8; ++v) rsum[v] = 1.0f / rsum[v];
    _Float16* ob = outT + (long)b * HW * DV;
#pragma unroll
    for (int ct = 0; ct < 8; ++ct)
#pragma unroll
        for (int v = 0; v < 8; ++v)
            ob[(long)(n0 + half * 8 + v) * DV + ct * 16 + l16] =
                (_Float16)(O[ct][v] * rsum[v]);
}

// ---------------------------------------------------------------------------
// Kernel 3: final conv+BN GEMM: y[o][n] = sum_c fin_wf[o][c]*outT[n][c] + t[o]
// ---------------------------------------------------------------------------
__global__ void final_gemm(const _Float16* __restrict__ outT, // [B][HW][128]
                           const _Float16* __restrict__ wf,   // [256][128]
                           const float* __restrict__ t,       // [256]
                           float* __restrict__ y) {           // [B][256][HW]
    const int lane = threadIdx.x & 31, wave = threadIdx.x >> 5;
    const int b = blockIdx.x, nT = blockIdx.y, og = blockIdx.z;
    const int n0 = nT * 16, o0 = og * 64 + wave * 16;
    const int half = lane >> 4, l16 = lane & 15;

    const _Float16* wrow = wf + (long)(o0 + l16) * DV;
    const _Float16* xrow = outT + ((long)b * HW + n0 + l16) * DV;

    v8f acc = {};
#pragma unroll
    for (int cb = 0; cb < DV; cb += 32) {
        v16h a  = ld8x2(wrow + cb + half * 16, wrow + cb + half * 16 + 8);
        v16h bf = ld16(xrow + cb + half * 16);
        acc = wmma16x16x32(a, bf, acc);
    }
#pragma unroll
    for (int v = 0; v < 8; ++v) {
        int o = o0 + half * 8 + v;
        y[((long)b * CC + o) * HW + n0 + l16] = acc[v] + t[o];
    }
}

// ---------------------------------------------------------------------------
// Host launcher
// ---------------------------------------------------------------------------
extern "C" void kernel_launch(void* const* d_in, const int* in_sizes, int n_in,
                              void* d_out, int out_size, void* d_ws, size_t ws_size,
                              hipStream_t stream) {
    (void)in_sizes; (void)n_in; (void)out_size; (void)ws_size;

    const float* x   = (const float*)d_in[0];
    const float* fea = (const float*)d_in[1];
    const float* const* th = (const float* const*)&d_in[2];   // theta w,b,g,be,m,v
    const float* const* ph = (const float* const*)&d_in[8];   // phi
    const float* const* gg = (const float* const*)&d_in[14];  // g
    const float* const* fi = (const float* const*)&d_in[20];  // fin
    float* y = (float*)d_out;

    // workspace carve-up (256B aligned slices); total ~26.3 MB
    size_t off = 0;
    auto take = [&](size_t bytes) -> void* {
        void* p = (char*)d_ws + off;
        off += (bytes + 255) & ~(size_t)255;
        return p;
    };
    _Float16* xh      = (_Float16*)take((size_t)NB * HW * CC * 2);
    _Float16* feah    = (_Float16*)take((size_t)NB * HW * CC * 2);
    _Float16* thetaT  = (_Float16*)take((size_t)NB * HW * DQK * 2);
    _Float16* phiT    = (_Float16*)take((size_t)NB * HW * DQK * 2);
    _Float16* gh      = (_Float16*)take((size_t)NB * DV * HW * 2);
    _Float16* outT    = (_Float16*)take((size_t)NB * HW * DV * 2);
    _Float16* theta_wf = (_Float16*)take((size_t)DQK * CC * 2);
    _Float16* phi_wf   = (_Float16*)take((size_t)DQK * CC * 2);
    _Float16* g_wf     = (_Float16*)take((size_t)DV * CC * 2);
    _Float16* fin_wf   = (_Float16*)take((size_t)CC * DV * 2);
    float* theta_t = (float*)take(DQK * 4);
    float* phi_t   = (float*)take(DQK * 4);
    float* g_t     = (float*)take(DV * 4);
    float* fin_t   = (float*)take(CC * 4);

    // 0a: BN folding
    fold_bn<<<(DQK * CC + 255) / 256, 256, 0, stream>>>(th[0], th[1], th[2], th[3], th[4], th[5], theta_wf, theta_t, DQK, CC);
    fold_bn<<<(DQK * CC + 255) / 256, 256, 0, stream>>>(ph[0], ph[1], ph[2], ph[3], ph[4], ph[5], phi_wf, phi_t, DQK, CC);
    fold_bn<<<(DV  * CC + 255) / 256, 256, 0, stream>>>(gg[0], gg[1], gg[2], gg[3], gg[4], gg[5], g_wf, g_t, DV, CC);
    fold_bn<<<(CC  * DV + 255) / 256, 256, 0, stream>>>(fi[0], fi[1], fi[2], fi[3], fi[4], fi[5], fin_wf, fin_t, CC, DV);

    // 0b: transpose+convert inputs to [B][HW][C] f16
    const long txe = (long)NB * CC * HW;
    xpose_h<<<(unsigned)((txe + 255) / 256), 256, 0, stream>>>(x, xh);
    xpose_h<<<(unsigned)((txe + 255) / 256), 256, 0, stream>>>(fea, feah);

    // 1: projections (WMMA GEMMs)
    proj_gemm<<<dim3(NB, HW / 64, DQK / 16), 128, 0, stream>>>(xh,   theta_wf, theta_t, thetaT, nullptr, DQK, 0);
    proj_gemm<<<dim3(NB, HW / 64, DQK / 16), 128, 0, stream>>>(feah, phi_wf,   phi_t,   phiT,   nullptr, DQK, 0);
    proj_gemm<<<dim3(NB, HW / 64, DV  / 16), 128, 0, stream>>>(feah, g_wf,     g_t,     nullptr, gh,     DV,  1);

    // 2: fused flash attention
    attn_kernel<<<dim3(NB, HW / 64), 128, 0, stream>>>(thetaT, phiT, gh, outT);

    // 3: final conv+BN GEMM -> f32 output
    final_gemm<<<dim3(NB, HW / 16, CC / 64), 128, 0, stream>>>(outT, fin_wf, fin_t, y);
}
